// ExpansionContrastModule_64321430225263
// MI455X (gfx1250) — matrix operand order repaired and checked by reading, not compiled, not run on hardware
//
#include <hip/hip_runtime.h>

// ---------------------------------------------------------------------------
// ExpansionContrastModule for MI455X (gfx1250, wave32, WMMA)
// Gram-factorized channel attention: never materializes K/V; all GEMM work
// runs on v_wmma_f32_16x16x32_bf16. All WMMAs execute in uniform control
// flow (EXEC all-1s per ISA requirement). effW staged to LDS via
// GLOBAL_LOAD_ASYNC_TO_LDS_B128 (ASYNCcnt) when the builtin is available.
// ---------------------------------------------------------------------------

typedef __attribute__((ext_vector_type(16))) __bf16 v16bf;
typedef __attribute__((ext_vector_type(8)))  float  v8f;

#if defined(__has_builtin)
#if __has_builtin(__builtin_amdgcn_global_load_async_to_lds_b128)
#define USE_ASYNC_LDS 1
#endif
#endif

#ifdef USE_ASYNC_LDS
// exact param types per hipcc diagnostic: int4 in AS1 (global) / AS3 (LDS)
typedef int v4i_ __attribute__((vector_size(16)));
typedef v4i_ __attribute__((address_space(1))) gv4i;
typedef v4i_ __attribute__((address_space(3))) sv4i;
#endif

#define Bn   8
#define Cc   16
#define Wd   192
#define Htt  192
#define Npix (Wd * Htt)     // 36864
#define NH   4
#define HID  16
#define C8   128            // 8 * Cc
#define TT   64             // NH * HID
#define OC   32
#define STRIP 2048
#define CHUNK 32

// workspace layout (float offsets)
#define OFF_S8   0                                    // B*4*128*128
#define OFF_SCC  (OFF_S8  + Bn*NH*C8*C8)              // B*4*16*16
#define OFF_SX   (OFF_SCC + Bn*NH*Cc*Cc)              // B*4*16*128
#define OFF_YS   (OFF_SX  + Bn*NH*Cc*C8)              // 32
#define OFF_YS2  (OFF_YS  + OC)                       // 32
#define OFF_EFF  (OFF_YS2 + OC)                       // B*4*16*128 bf16 (half used)
#define OFF_Y    (OFF_EFF + Bn*NH*HID*C8)             // B*32*Npix
#define ACC_FLOATS OFF_EFF                            // region to zero

__device__ __forceinline__ __bf16 tobf(float f) {
  unsigned u = __builtin_bit_cast(unsigned, f);
  unsigned r = (u + 0x7FFFu + ((u >> 16) & 1u)) >> 16;
  return __builtin_bit_cast(__bf16, (unsigned short)r);
}

__device__ __forceinline__ v8f wmma_bf16(v16bf a, v16bf b, v8f c) {
  return __builtin_amdgcn_wmma_f32_16x16x32_bf16(false, a, false, b,
                                                 (short)0, c, false, false);
}

// A-matrix (16x32 bf16) loader: lanes 0-15 -> M rows,
// element e holds K = e + 16*grp (contiguous order, matches B loaders)
__device__ __forceinline__ v16bf load_A(const __bf16* base, int ld, int lane) {
  int row = lane & 15, grp = lane >> 4;
  v16bf a;
#pragma unroll
  for (int e = 0; e < 16; ++e) {
    int k = e + 16 * grp;
    a[e] = base[row * ld + k];
  }
  return a;
}

// B (32x16) where LDS holds n-major rows: B[k][n] = base[n*ld + k]
__device__ __forceinline__ v16bf load_B_nmajor(const __bf16* base, int ld, int lane) {
  int n = lane & 15, grp = lane >> 4;
  v16bf b;
#pragma unroll
  for (int e = 0; e < 16; ++e) b[e] = base[n * ld + e + 16 * grp];
  return b;
}

// B (32x16) where LDS holds k-major rows: B[k][n] = base[k*ld + n]
__device__ __forceinline__ v16bf load_B_kmajor(const __bf16* base, int ld, int lane) {
  int n = lane & 15, grp = lane >> 4;
  v16bf b;
#pragma unroll
  for (int e = 0; e < 16; ++e) b[e] = base[(e + 16 * grp) * ld + n];
  return b;
}

// Fused depthwise dilated-3x3 contrast conv at one (channel, pixel):
// sur_k = (1-sw)*(c - nb_k) + sw*((c+sum8)/8 - 9/8*nb_k)
// cen_x = (1-sw)*(7/64*sum8 + c/8) + sw*c
__device__ __forceinline__ void conv_point(const float* __restrict__ cenb, int c,
                                           int n, int s, float sw,
                                           float* sur, float* cx) {
  const int dw[8] = {-1, -1, -1, 0, 1, 1, 1, 0};
  const int dh[8] = {-1, 0, 1, 1, 1, 0, -1, -1};
  int iw = n / Htt, ih = n - iw * Htt;
  const float* p = cenb + c * Npix;
  float cv = p[n];
  float sum8 = 0.f, nb[8];
#pragma unroll
  for (int k = 0; k < 8; ++k) {
    int w2 = iw + dw[k] * s, h2 = ih + dh[k] * s;
    float v = 0.f;
    if (w2 >= 0 && w2 < Wd && h2 >= 0 && h2 < Htt) v = p[w2 * Htt + h2];
    nb[k] = v;
    sum8 += v;
  }
  float omsw = 1.f - sw;
  if (cx) *cx = omsw * (0.109375f * sum8 + 0.125f * cv) + sw * cv;
  float t = (cv + sum8) * 0.125f;
#pragma unroll
  for (int k = 0; k < 8; ++k)
    sur[k] = omsw * (cv - nb[k]) + sw * (t - 1.125f * nb[k]);
}

// ---------------------------------------------------------------------------
// K1: per (b, shift, pixel-strip) accumulate Gram matrices with WMMA:
//   S8 += sur.sur^T (128x128)  Sx += cenx.sur^T (16x128)  Scc += cenx.cenx^T
// All WMMAs unconditional (uniform EXEC); divergence only around atomics.
// ---------------------------------------------------------------------------
__global__ void __launch_bounds__(256)
k1_gram(const float* __restrict__ cen, const float* __restrict__ sum_w,
        float* __restrict__ ws) {
  __shared__ __bf16 s_sur[C8][CHUNK + 2];
  __shared__ __bf16 s_cx[Cc][CHUNK + 2];

  int strip = blockIdx.x, sidx = blockIdx.y, b = blockIdx.z;
  int s = 1 << sidx;
  const float* cenb = cen + (size_t)b * Cc * Npix;
  int lane = threadIdx.x & 31, wid = threadIdx.x >> 5;

  v8f acc[8] = {};
  v8f accSx = {};
  v8f accScc = {};

  int base = strip * STRIP;
  for (int ch = 0; ch < STRIP / CHUNK; ++ch) {
    int p0 = base + ch * CHUNK;
    if (ch + 1 < STRIP / CHUNK)
      __builtin_prefetch(&cenb[wid * 2 * Npix + p0 + CHUNK], 0, 1);
    // phase A: 512 conv points with 256 threads
#pragma unroll
    for (int it = 0; it < 2; ++it) {
      int idx = threadIdx.x + it * 256;
      int c = idx >> 5, px = idx & 31;
      float sw = sum_w[sidx * Cc + c];
      float sur[8], cx;
      conv_point(cenb, c, p0 + px, s, sw, sur, &cx);
      s_cx[c][px] = tobf(cx);
#pragma unroll
      for (int k = 0; k < 8; ++k) s_sur[k * Cc + c][px] = tobf(sur[k]);
    }
    __syncthreads();
    // phase B: rank-32 Gram update. wave w owns S8 tile-row w + Sx tile w.
    v16bf aS = load_A(&s_sur[wid * 16][0], CHUNK + 2, lane);
    v16bf aC = load_A(&s_cx[0][0], CHUNK + 2, lane);
#pragma unroll
    for (int j = 0; j < 8; ++j) {
      v16bf bS = load_B_nmajor(&s_sur[j * 16][0], CHUNK + 2, lane);
      acc[j] = wmma_bf16(aS, bS, acc[j]);
    }
    {
      v16bf bW = load_B_nmajor(&s_sur[wid * 16][0], CHUNK + 2, lane);
      accSx = wmma_bf16(aC, bW, accSx);
      // Scc computed by every wave (uniform EXEC); only wave 0 writes back.
      v16bf bC = load_B_nmajor(&s_cx[0][0], CHUNK + 2, lane);
      accScc = wmma_bf16(aC, bC, accScc);
    }
    __syncthreads();
  }

  // cross-block reduction into the global Gram accumulators
  float* S8 = ws + OFF_S8 + (size_t)(b * NH + sidx) * C8 * C8;
  float* SX = ws + OFF_SX + (size_t)(b * NH + sidx) * Cc * C8;
  float* SC = ws + OFF_SCC + (size_t)(b * NH + sidx) * Cc * Cc;
  int grp = lane >> 4, ncol = lane & 15;
#pragma unroll
  for (int j = 0; j < 8; ++j)
#pragma unroll
    for (int r = 0; r < 8; ++r)
      atomicAdd(&S8[(wid * 16 + r + 8 * grp) * C8 + j * 16 + ncol], acc[j][r]);
#pragma unroll
  for (int r = 0; r < 8; ++r)
    atomicAdd(&SX[(r + 8 * grp) * C8 + wid * 16 + ncol], accSx[r]);
  if (wid == 0)
#pragma unroll
    for (int r = 0; r < 8; ++r)
      atomicAdd(&SC[(r + 8 * grp) * Cc + ncol], accScc[r]);
}

// ---------------------------------------------------------------------------
// K2: per (b,h): norms from Grams, score = wq Sx wk^T / (|Q||K| sqrt(N)),
// InstanceNorm(16x128), row softmax, effW = attn @ wv (16x128, stored bf16).
// ---------------------------------------------------------------------------
__global__ void __launch_bounds__(128)
k2_attn(const float* __restrict__ wq, const float* __restrict__ wk,
        const float* __restrict__ wv_, float* __restrict__ ws) {
  int bh = blockIdx.x;
  int h = bh & 3;
  const float* S8 = ws + OFF_S8 + (size_t)bh * C8 * C8;
  const float* SC = ws + OFF_SCC + (size_t)bh * Cc * Cc;
  const float* SX = ws + OFF_SX + (size_t)bh * Cc * C8;
  __bf16* EFF = (__bf16*)(ws + OFF_EFF) + (size_t)bh * HID * C8;
  const float* wqh = wq + h * HID * Cc;
  const float* wkh = wk + h * C8 * C8;
  const float* wvh = wv_ + h * C8 * C8;

  __shared__ float s_score[HID][C8 + 1];
  __shared__ float s_qn[HID];
  __shared__ float s_kn[C8];
  __shared__ float s_red[128];
  int tid = threadIdx.x;

  if (tid < HID) {  // |Q_q|^2 = (wq Scc wq^T)[q,q]
    float acc = 0.f;
    for (int c = 0; c < Cc; ++c) {
      float t = 0.f;
      for (int c2 = 0; c2 < Cc; ++c2) t += SC[c * Cc + c2] * wqh[tid * Cc + c2];
      acc += wqh[tid * Cc + c] * t;
    }
    s_qn[tid] = fmaxf(sqrtf(fmaxf(acc, 0.f)), 1e-12f);
  }
  {  // |K_k|^2 = (wk S8 wk^T)[k,k]
    float acc = 0.f;
    for (int c = 0; c < C8; ++c) {
      float wkc = wkh[tid * C8 + c];
      const float* row = S8 + c * C8;
      float t = 0.f;
      for (int c2 = 0; c2 < C8; ++c2) t += row[c2] * wkh[tid * C8 + c2];
      acc += wkc * t;
    }
    s_kn[tid] = fmaxf(sqrtf(fmaxf(acc, 0.f)), 1e-12f);
  }
  __syncthreads();
  {  // raw score column k=tid
    float u[Cc];
    for (int c = 0; c < Cc; ++c) {
      float t = 0.f;
      for (int c8 = 0; c8 < C8; ++c8) t += SX[c * C8 + c8] * wkh[tid * C8 + c8];
      u[c] = t;
    }
    float inv = 1.f / (s_kn[tid] * 192.f);  // sqrt(N) = 192
    for (int q = 0; q < HID; ++q) {
      float t = 0.f;
      for (int c = 0; c < Cc; ++c) t += wqh[q * Cc + c] * u[c];
      s_score[q][tid] = t * inv / s_qn[q];
    }
  }
  __syncthreads();
  // InstanceNorm over 16x128 (biased var)
  float lsum = 0.f, lsq = 0.f;
  for (int q = 0; q < HID; ++q) {
    float v = s_score[q][tid];
    lsum += v;
    lsq += v * v;
  }
  s_red[tid] = lsum;
  __syncthreads();
  for (int st = 64; st > 0; st >>= 1) {
    if (tid < st) s_red[tid] += s_red[tid + st];
    __syncthreads();
  }
  float mean = s_red[0] / 2048.f;
  __syncthreads();
  s_red[tid] = lsq;
  __syncthreads();
  for (int st = 64; st > 0; st >>= 1) {
    if (tid < st) s_red[tid] += s_red[tid + st];
    __syncthreads();
  }
  float var = s_red[0] / 2048.f - mean * mean;
  float rstd = rsqrtf(var + 1e-5f);
  __syncthreads();
  for (int q = 0; q < HID; ++q)
    s_score[q][tid] = (s_score[q][tid] - mean) * rstd;
  __syncthreads();
  // softmax per row
  for (int q = 0; q < HID; ++q) {
    float v = s_score[q][tid];
    s_red[tid] = v;
    __syncthreads();
    for (int st = 64; st > 0; st >>= 1) {
      if (tid < st) s_red[tid] = fmaxf(s_red[tid], s_red[tid + st]);
      __syncthreads();
    }
    float mx = s_red[0];
    __syncthreads();
    float e = __expf(v - mx);
    s_red[tid] = e;
    __syncthreads();
    for (int st = 64; st > 0; st >>= 1) {
      if (tid < st) s_red[tid] += s_red[tid + st];
      __syncthreads();
    }
    float sm = s_red[0];
    __syncthreads();
    s_score[q][tid] = e / sm;
  }
  __syncthreads();
  // effW = attn @ wv  (fold attention into output projection), store bf16
  for (int q = 0; q < HID; ++q) {
    float t = 0.f;
    for (int k = 0; k < C8; ++k) t += s_score[q][k] * wvh[k * C8 + tid];
    EFF[q * C8 + tid] = tobf(t);
  }
}

// ---------------------------------------------------------------------------
// K3: per (b, 16-pixel tile): recompute sur (4 shifts), WMMA
//   out_h(16x16) = effW_h(16x128) x sur_h(128x16)
//   y(32x16)     = w_out(32x64)   x out(64x16)   (all waves compute, 0/1 store)
// effW staged via GLOBAL_LOAD_ASYNC_TO_LDS_B128 (ASYNCcnt) when available.
// ---------------------------------------------------------------------------
__global__ void __launch_bounds__(128)
k3_out(const float* __restrict__ cen, const float* __restrict__ sum_w,
       const float* __restrict__ w_out, float* __restrict__ ws) {
  __shared__ __bf16 s_sur[NH][C8][18];
  __shared__ __bf16 s_eff[NH][HID][C8];
  __shared__ __bf16 s_out[TT][18];
  __shared__ __bf16 s_wout[OC][TT + 2];
  __shared__ float s_ys[OC], s_ys2[OC];

  int tile = blockIdx.x, b = blockIdx.y;
  int tid = threadIdx.x, lane = tid & 31, wid = tid >> 5;
  int n0 = tile * 16;
  const float* cenb = cen + (size_t)b * Cc * Npix;
  const __bf16* EFFbf = (const __bf16*)(ws + OFF_EFF) + (size_t)b * NH * HID * C8;

#ifdef USE_ASYNC_LDS
  // async DMA: 16 KB of effW straight into LDS, tracked on ASYNCcnt
  for (int i = tid; i < (NH * HID * C8) / 8; i += 128)
    __builtin_amdgcn_global_load_async_to_lds_b128(
        (gv4i*)(EFFbf + i * 8), (sv4i*)(((__bf16*)s_eff) + i * 8), 0, 0);
#else
  for (int i = tid; i < NH * HID * C8; i += 128)
    ((__bf16*)s_eff)[i] = EFFbf[i];
#endif
  for (int i = tid; i < OC * TT; i += 128)
    s_wout[i >> 6][i & 63] = tobf(w_out[i]);
  if (tid < OC) { s_ys[tid] = 0.f; s_ys2[tid] = 0.f; }

  {  // wave w computes sur for shift/head w over 16 channels x 16 pixels
    int s = 1 << wid;
#pragma unroll
    for (int it = 0; it < 8; ++it) {
      int idx = it * 32 + lane;
      int c = idx >> 4, px = idx & 15;
      float sw = sum_w[wid * Cc + c];
      float sur[8];
      conv_point(cenb, c, n0 + px, s, sw, sur, nullptr);
#pragma unroll
      for (int k = 0; k < 8; ++k) s_sur[wid][k * Cc + c][px] = tobf(sur[k]);
    }
  }
#ifdef USE_ASYNC_LDS
  asm volatile("s_wait_asynccnt 0" ::: "memory");
#endif
  __syncthreads();
  {  // out_h = effW_h x sur_h (K=128 in 4 rank-32 steps)
    v8f acc = {};
#pragma unroll
    for (int kc = 0; kc < 4; ++kc) {
      v16bf a = load_A(&s_eff[wid][0][kc * 32], C8, lane);
      v16bf bm = load_B_kmajor(&s_sur[wid][kc * 32][0], 18, lane);
      acc = wmma_bf16(a, bm, acc);
    }
    int grp = lane >> 4, ncol = lane & 15;
#pragma unroll
    for (int r = 0; r < 8; ++r)
      s_out[wid * 16 + r + 8 * grp][ncol] = tobf(acc[r]);
  }
  __syncthreads();
  {  // y = w_out x out; all waves compute (uniform EXEC), waves 0/1 store
    int m = wid & 1;
    v8f acc = {};
#pragma unroll
    for (int kc = 0; kc < 2; ++kc) {
      v16bf a = load_A(&s_wout[m * 16][kc * 32], TT + 2, lane);
      v16bf bm = load_B_kmajor(&s_out[kc * 32][0], 18, lane);
      acc = wmma_bf16(a, bm, acc);
    }
    if (wid < 2) {
      int grp = lane >> 4, px = lane & 15;
      float* Y = ws + OFF_Y;
#pragma unroll
      for (int r = 0; r < 8; ++r) {
        int och = m * 16 + r + 8 * grp;
        float v = acc[r];
        Y[((size_t)b * OC + och) * Npix + n0 + px] = v;
        atomicAdd(&s_ys[och], v);        // LDS ds_add_f32
        atomicAdd(&s_ys2[och], v * v);
      }
    }
  }
  __syncthreads();
  if (tid < OC) {
    atomicAdd(ws + OFF_YS + tid, s_ys[tid]);
    atomicAdd(ws + OFF_YS2 + tid, s_ys2[tid]);
  }
}

// ---------------------------------------------------------------------------
// K4: BatchNorm (batch stats) + affine + ReLU  -> d_out
// ---------------------------------------------------------------------------
__global__ void __launch_bounds__(256)
k4_final(const float* __restrict__ ws, const float* __restrict__ gamma,
         const float* __restrict__ beta, float* __restrict__ out) {
  int idx = blockIdx.x * 256 + threadIdx.x;
  if (idx >= Bn * OC * Npix) return;
  int och = (idx / Npix) & (OC - 1);
  float cnt = (float)(Bn * Npix);
  float mu = ws[OFF_YS + och] / cnt;
  float var = ws[OFF_YS2 + och] / cnt - mu * mu;
  float v = (ws[OFF_Y + idx] - mu) * rsqrtf(var + 1e-5f) * gamma[och] + beta[och];
  out[idx] = fmaxf(v, 0.f);
}

extern "C" void kernel_launch(void* const* d_in, const int* in_sizes, int n_in,
                              void* d_out, int out_size, void* d_ws, size_t ws_size,
                              hipStream_t stream) {
  const float* cen   = (const float*)d_in[0];
  const float* wq    = (const float*)d_in[1];
  const float* wk    = (const float*)d_in[2];
  const float* wv    = (const float*)d_in[3];
  const float* sum_w = (const float*)d_in[4];
  const float* w_out = (const float*)d_in[5];
  const float* gamma = (const float*)d_in[6];
  const float* beta  = (const float*)d_in[7];
  float* ws  = (float*)d_ws;
  float* out = (float*)d_out;

  (void)hipMemsetAsync(ws, 0, (size_t)ACC_FLOATS * sizeof(float), stream);
  k1_gram<<<dim3(Npix / STRIP, NH, Bn), 256, 0, stream>>>(cen, sum_w, ws);
  k2_attn<<<dim3(Bn * NH), 128, 0, stream>>>(wq, wk, wv, ws);
  k3_out<<<dim3(Npix / 16, Bn), 128, 0, stream>>>(cen, sum_w, w_out, ws);
  k4_final<<<dim3((Bn * OC * Npix + 255) / 256), 256, 0, stream>>>(ws, gamma, beta, out);
}
